// NeuronLTX23TransformerBackbone_64046552318362
// MI455X (gfx1250) — matrix operand order repaired and checked by reading, not compiled, not run on hardware
//
#include <hip/hip_runtime.h>
#include <hip/hip_bf16.h>
#include <math.h>

#define DIM   4096
#define SEQ   4096
#define TSEQ  256
#define HEADS 32
#define HD    128
#define FFN   16384

typedef __bf16 bf16;
typedef __attribute__((ext_vector_type(16))) __bf16 v16bf;
typedef __attribute__((ext_vector_type(8)))  __bf16 v8bf;
typedef __attribute__((ext_vector_type(8)))  float  v8f;

static __device__ __forceinline__ v16bf cat16(v8bf lo, v8bf hi) {
  return __builtin_shufflevector(lo, hi, 0,1,2,3,4,5,6,7,8,9,10,11,12,13,14,15);
}
static __device__ __forceinline__ unsigned pk(float x, float y) {
  union { bf16 h[2]; unsigned u; } u;
  u.h[0] = (bf16)x; u.h[1] = (bf16)y;
  return u.u;
}
static __device__ __forceinline__ float f4g(const float4& v, int i) {
  return ((const float*)&v)[i];
}
// branchless tanh-GELU: tanh(u) = 1 - 2/(exp(2u)+1)  (safe at +-inf)
static __device__ __forceinline__ float gelu_tanh(float g) {
  float u  = 0.7978845608028654f * (g + 0.044715f * g * g * g);
  float e  = __expf(2.0f * u);
  float th = 1.0f - 2.0f / (e + 1.0f);
  return 0.5f * g * (1.0f + th);
}

// ---------------------------------------------------------------------------
// Generic GEMM: C[M,N] = A[M,K] @ W[K,N]  (+ optional residual R)
// fp32 in/out, bf16 WMMA compute. Tile 128x128x32, 8 waves, wave tile 64x32.
// Software pipelined: next tile's global loads staged in regs during WMMA.
// ---------------------------------------------------------------------------
template<int ADD_RES>
__global__ __launch_bounds__(256)
void gemm_bf16(const float* __restrict__ A, const float* __restrict__ W,
               const float* __restrict__ R, float* __restrict__ C,
               int M, int N, int K)
{
  __shared__ __align__(16) bf16 As[128][32];
  __shared__ __align__(16) bf16 Bs[128][32];   // transposed: Bs[n][k]

  const int t    = threadIdx.x;
  const int lane = t & 31;
  const int w    = t >> 5;
  const int hl   = lane & 15;
  const int hh   = lane >> 4;
  const int m0   = (w & 1) * 64;
  const int n0   = (w >> 1) * 32;
  const int tileM = blockIdx.y * 128;
  const int tileN = blockIdx.x * 128;

  v8f zero = {};
  v8f acc[4][2];
  for (int i = 0; i < 4; ++i)
    for (int j = 0; j < 2; ++j) acc[i][j] = zero;

  const int arow = t >> 1;          // 0..127
  const int acol = (t & 1) * 16;    // 0 / 16
  const int bkp  = (t & 15) * 2;    // even k row 0..30
  const int bc   = (t >> 4) * 8;    // col group 0..120

  const float* aptr = A + (size_t)(tileM + arow) * K + acol;
  const float* bptr = W + (size_t)bkp * N + tileN + bc;

  float4 ra[4], rb[4];
#pragma unroll
  for (int j = 0; j < 4; ++j) ra[j] = *(const float4*)(aptr + 4 * j);
  rb[0] = *(const float4*)(bptr);
  rb[1] = *(const float4*)(bptr + 4);
  rb[2] = *(const float4*)(bptr + N);
  rb[3] = *(const float4*)(bptr + N + 4);

  for (int k0 = 0; k0 < K; k0 += 32) {
    __syncthreads();                 // prior compute done reading LDS
    // ---- commit staged regs to LDS (packed bf16) ----
#pragma unroll
    for (int j = 0; j < 4; ++j)
      *(uint2*)&As[arow][acol + 4 * j] =
          make_uint2(pk(ra[j].x, ra[j].y), pk(ra[j].z, ra[j].w));
#pragma unroll
    for (int j = 0; j < 4; ++j) {
      *(unsigned*)&Bs[bc + j][bkp]     = pk(f4g(rb[0], j), f4g(rb[2], j));
      *(unsigned*)&Bs[bc + 4 + j][bkp] = pk(f4g(rb[1], j), f4g(rb[3], j));
    }
    __syncthreads();

    // ---- stage next k tile (latency overlapped with WMMA below) ----
    if (k0 + 32 < K) {
      const float* an = aptr + k0 + 32;
      const float* bn = W + (size_t)(k0 + 32 + bkp) * N + tileN + bc;
#pragma unroll
      for (int j = 0; j < 4; ++j) ra[j] = *(const float4*)(an + 4 * j);
      rb[0] = *(const float4*)(bn);
      rb[1] = *(const float4*)(bn + 4);
      rb[2] = *(const float4*)(bn + N);
      rb[3] = *(const float4*)(bn + N + 4);
    }

    // ---- fragments + WMMA ----
    v16bf afrag[4];
#pragma unroll
    for (int mt = 0; mt < 4; ++mt) {
      int r  = m0 + mt * 16 + hl;
      int kb = hh * 8;
      afrag[mt] = cat16(*(const v8bf*)&As[r][kb], *(const v8bf*)&As[r][kb + 16]);
    }
    v16bf bfrag[2];
#pragma unroll
    for (int nt = 0; nt < 2; ++nt) {
      int c  = n0 + nt * 16 + hl;
      int kl = hh * 8;
      bfrag[nt] = cat16(*(const v8bf*)&Bs[c][kl], *(const v8bf*)&Bs[c][kl + 16]);
    }
#pragma unroll
    for (int mt = 0; mt < 4; ++mt)
#pragma unroll
      for (int nt = 0; nt < 2; ++nt)
        acc[mt][nt] = __builtin_amdgcn_wmma_f32_16x16x32_bf16(
            false, afrag[mt], false, bfrag[nt], (short)0, acc[mt][nt], false, false);
  }

#pragma unroll
  for (int mt = 0; mt < 4; ++mt)
#pragma unroll
    for (int nt = 0; nt < 2; ++nt)
#pragma unroll
      for (int i = 0; i < 8; ++i) {
        int row = tileM + m0 + mt * 16 + i + 8 * hh;
        int col = tileN + n0 + nt * 16 + hl;
        float v = acc[mt][nt][i];
        if (ADD_RES) v += R[(size_t)row * N + col];
        C[(size_t)row * N + col] = v;
      }
}

// ---------------------------------------------------------------------------
// GEGLU proj GEMM: computes val = A@W[:, j] and gate = A@W[:, FFN+j] for the
// same col block, stores val * gelu(gate). Tile 128x64x32, wave tile 32x32.
// blockIdx.x = M blocks (fast) so concurrent blocks share the weight columns
// (512MB W_proj streams HBM once; 64MB activation stays L2-resident).
// ---------------------------------------------------------------------------
__global__ __launch_bounds__(256)
void gemm_geglu(const float* __restrict__ A, const float* __restrict__ W,
                float* __restrict__ G, int M, int K)
{
  __shared__ __align__(16) bf16 As[128][32];
  __shared__ __align__(16) bf16 Bv[64][32];
  __shared__ __align__(16) bf16 Bg[64][32];

  const int t    = threadIdx.x;
  const int lane = t & 31;
  const int w    = t >> 5;
  const int hl   = lane & 15;
  const int hh   = lane >> 4;
  const int m0   = (w & 3) * 32;
  const int n0   = (w >> 2) * 32;
  const int tileM = blockIdx.x * 128;   // M on fast axis
  const int ng0   = blockIdx.y * 64;    // col base in FFN space
  const int ldw   = 2 * FFN;

  v8f zero = {};
  v8f accV[2][2], accG[2][2];
  for (int i = 0; i < 2; ++i)
    for (int j = 0; j < 2; ++j) { accV[i][j] = zero; accG[i][j] = zero; }

  const int arow = t >> 1;
  const int acol = (t & 1) * 16;
  const int bkp  = (t & 15) * 2;    // even k row
  const int bc   = (t >> 4) * 4;    // col group 0..60

  const float* aptr = A + (size_t)(tileM + arow) * K + acol;
  const float* bbase = W + ng0 + bc;

  float4 ra[4], rv[2], rg[2];
#pragma unroll
  for (int j = 0; j < 4; ++j) ra[j] = *(const float4*)(aptr + 4 * j);
  rv[0] = *(const float4*)(bbase + (size_t)bkp * ldw);
  rv[1] = *(const float4*)(bbase + (size_t)(bkp + 1) * ldw);
  rg[0] = *(const float4*)(bbase + (size_t)bkp * ldw + FFN);
  rg[1] = *(const float4*)(bbase + (size_t)(bkp + 1) * ldw + FFN);

  for (int k0 = 0; k0 < K; k0 += 32) {
    __syncthreads();
#pragma unroll
    for (int j = 0; j < 4; ++j)
      *(uint2*)&As[arow][acol + 4 * j] =
          make_uint2(pk(ra[j].x, ra[j].y), pk(ra[j].z, ra[j].w));
#pragma unroll
    for (int j = 0; j < 4; ++j) {
      *(unsigned*)&Bv[bc + j][bkp] = pk(f4g(rv[0], j), f4g(rv[1], j));
      *(unsigned*)&Bg[bc + j][bkp] = pk(f4g(rg[0], j), f4g(rg[1], j));
    }
    __syncthreads();

    if (k0 + 32 < K) {
      const float* an = aptr + k0 + 32;
      const float* bn = bbase + (size_t)(k0 + 32 + bkp) * ldw;
#pragma unroll
      for (int j = 0; j < 4; ++j) ra[j] = *(const float4*)(an + 4 * j);
      rv[0] = *(const float4*)(bn);
      rv[1] = *(const float4*)(bn + ldw);
      rg[0] = *(const float4*)(bn + FFN);
      rg[1] = *(const float4*)(bn + ldw + FFN);
    }

    v16bf afrag[2];
#pragma unroll
    for (int mt = 0; mt < 2; ++mt) {
      int r  = m0 + mt * 16 + hl;
      int kb = hh * 8;
      afrag[mt] = cat16(*(const v8bf*)&As[r][kb], *(const v8bf*)&As[r][kb + 16]);
    }
#pragma unroll
    for (int nt = 0; nt < 2; ++nt) {
      int c  = n0 + nt * 16 + hl;
      int kl = hh * 8;
      v16bf bfv = cat16(*(const v8bf*)&Bv[c][kl], *(const v8bf*)&Bv[c][kl + 16]);
      v16bf bfg = cat16(*(const v8bf*)&Bg[c][kl], *(const v8bf*)&Bg[c][kl + 16]);
#pragma unroll
      for (int mt = 0; mt < 2; ++mt) {
        accV[mt][nt] = __builtin_amdgcn_wmma_f32_16x16x32_bf16(
            false, afrag[mt], false, bfv, (short)0, accV[mt][nt], false, false);
        accG[mt][nt] = __builtin_amdgcn_wmma_f32_16x16x32_bf16(
            false, afrag[mt], false, bfg, (short)0, accG[mt][nt], false, false);
      }
    }
  }

#pragma unroll
  for (int mt = 0; mt < 2; ++mt)
#pragma unroll
    for (int nt = 0; nt < 2; ++nt)
#pragma unroll
      for (int i = 0; i < 8; ++i) {
        int row = tileM + m0 + mt * 16 + i + 8 * hh;
        int col = ng0 + n0 + nt * 16 + hl;
        G[(size_t)row * FFN + col] = accV[mt][nt][i] * gelu_tanh(accG[mt][nt][i]);
      }
}

// ---------------------------------------------------------------------------
// Flash attention. Grid (Sq/64, HEADS), 128 threads = 4 waves, each wave
// owns 16 query rows. K/V^T staged in LDS as packed bf16; QK^T and PV via
// WMMA; online softmax with per-row stats in LDS.
// ---------------------------------------------------------------------------
__global__ __launch_bounds__(128)
void flash_attn(const float* __restrict__ Q, const float* __restrict__ Kmat,
                const float* __restrict__ V, float* __restrict__ O, int Skv)
{
  __shared__ __align__(16) bf16  Ks[64][HD];      // keys row-major
  __shared__ __align__(16) bf16  Vt[HD][64];      // V transposed: [hd][key]
  __shared__ __align__(16) float Pw[4][16][64];   // per-wave prob tile
  __shared__ float rm[4][16], rl[4][16], rs[4][16];

  const int t    = threadIdx.x;
  const int lane = t & 31;
  const int w    = t >> 5;
  const int hl   = lane & 15;
  const int hh   = lane >> 4;
  const int h    = blockIdx.y;
  const int qb   = blockIdx.x * 64;
  const float scale = 0.08838834764831845f;  // 1/sqrt(128)

  // ---- load Q fragments for this wave's 16 rows (kept in VGPRs) ----
  v16bf aq[4];
  {
    const float* qrow = Q + (size_t)(qb + w * 16 + hl) * DIM + h * HD;
#pragma unroll
    for (int ks = 0; ks < 4; ++ks) {
      int kb = ks * 32 + hh * 8;
      float4 f0 = *(const float4*)(qrow + kb);
      float4 f1 = *(const float4*)(qrow + kb + 4);
      float4 f2 = *(const float4*)(qrow + kb + 16);
      float4 f3 = *(const float4*)(qrow + kb + 20);
      v16bf a;
      a[0]=(bf16)f0.x; a[1]=(bf16)f0.y; a[2]=(bf16)f0.z; a[3]=(bf16)f0.w;
      a[4]=(bf16)f1.x; a[5]=(bf16)f1.y; a[6]=(bf16)f1.z; a[7]=(bf16)f1.w;
      a[8]=(bf16)f2.x; a[9]=(bf16)f2.y; a[10]=(bf16)f2.z; a[11]=(bf16)f2.w;
      a[12]=(bf16)f3.x; a[13]=(bf16)f3.y; a[14]=(bf16)f3.z; a[15]=(bf16)f3.w;
      aq[ks] = a;
    }
  }

  v8f zero = {};
  v8f oacc[8];
  for (int i = 0; i < 8; ++i) oacc[i] = zero;
  if (lane < 16) { rm[w][lane] = -1e30f; rl[w][lane] = 0.0f; }

  for (int kb0 = 0; kb0 < Skv; kb0 += 64) {
    __syncthreads();
    // ---- stage 64 keys (packed b64 stores) ----
    {
      int kr = t >> 1, cb = (t & 1) * 64;
      const float* src = Kmat + (size_t)(kb0 + kr) * DIM + h * HD + cb;
#pragma unroll
      for (int j = 0; j < 64; j += 4) {
        float4 f = *(const float4*)(src + j);
        *(uint2*)&Ks[kr][cb + j] = make_uint2(pk(f.x, f.y), pk(f.z, f.w));
      }
    }
    // ---- stage V transposed, two keys packed per dword ----
    {
      int kp = (lane << 1) | 0;        // even key, 0..62 per wave? use thread mapping:
      kp = (t & 31) * 2;
      int hb = (t >> 5) * 32;
      const float* s0 = V + (size_t)(kb0 + kp) * DIM + h * HD + hb;
      const float* s1 = s0 + DIM;
#pragma unroll
      for (int j = 0; j < 32; j += 4) {
        float4 a = *(const float4*)(s0 + j);
        float4 b = *(const float4*)(s1 + j);
        *(unsigned*)&Vt[hb + j + 0][kp] = pk(a.x, b.x);
        *(unsigned*)&Vt[hb + j + 1][kp] = pk(a.y, b.y);
        *(unsigned*)&Vt[hb + j + 2][kp] = pk(a.z, b.z);
        *(unsigned*)&Vt[hb + j + 3][kp] = pk(a.w, b.w);
      }
    }
    __syncthreads();

    // ---- scores: S = Q(16x128) @ K^T(128x64) ----
#pragma unroll
    for (int nt = 0; nt < 4; ++nt) {
      v8f c = zero;
#pragma unroll
      for (int ks = 0; ks < 4; ++ks) {
        int krow = nt * 16 + hl;
        int kl   = ks * 32 + hh * 8;
        v16bf bk = cat16(*(const v8bf*)&Ks[krow][kl], *(const v8bf*)&Ks[krow][kl + 16]);
        c = __builtin_amdgcn_wmma_f32_16x16x32_bf16(
            false, aq[ks], false, bk, (short)0, c, false, false);
      }
#pragma unroll
      for (int i = 0; i < 8; ++i)
        Pw[w][i + 8 * hh][nt * 16 + hl] = c[i] * scale;
    }

    // ---- online softmax (lanes 0..15, one row each) ----
    if (lane < 16) {
      int r = lane;
      float oldm = rm[w][r];
      float bm = oldm;
      for (int j = 0; j < 64; ++j) bm = fmaxf(bm, Pw[w][r][j]);
      float alpha = __expf(oldm - bm);
      float s = 0.0f;
      for (int j = 0; j < 64; ++j) {
        float p = __expf(Pw[w][r][j] - bm);
        Pw[w][r][j] = p;
        s += p;
      }
      rm[w][r] = bm;
      rl[w][r] = rl[w][r] * alpha + s;
      rs[w][r] = alpha;
    }

    // ---- rescale running output ----
    float al[8];
#pragma unroll
    for (int i = 0; i < 8; ++i) al[i] = rs[w][i + 8 * hh];
#pragma unroll
    for (int nt = 0; nt < 8; ++nt)
#pragma unroll
      for (int i = 0; i < 8; ++i) oacc[nt][i] *= al[i];

    // ---- PV: O += P(16x64) @ V(64x128) ----
#pragma unroll
    for (int ks = 0; ks < 2; ++ks) {
      v16bf ap;
      int kb = ks * 32 + hh * 8;
#pragma unroll
      for (int e = 0; e < 8; ++e) ap[e]     = (bf16)Pw[w][hl][kb + e];
#pragma unroll
      for (int e = 0; e < 8; ++e) ap[8 + e] = (bf16)Pw[w][hl][kb + 16 + e];
#pragma unroll
      for (int nt = 0; nt < 8; ++nt) {
        int hr = nt * 16 + hl;
        int kl = ks * 32 + hh * 8;
        v16bf bv = cat16(*(const v8bf*)&Vt[hr][kl], *(const v8bf*)&Vt[hr][kl + 16]);
        oacc[nt] = __builtin_amdgcn_wmma_f32_16x16x32_bf16(
            false, ap, false, bv, (short)0, oacc[nt], false, false);
      }
    }
  }

  // ---- normalize + store ----
  float inv[8];
#pragma unroll
  for (int i = 0; i < 8; ++i) inv[i] = 1.0f / rl[w][i + 8 * hh];
#pragma unroll
  for (int nt = 0; nt < 8; ++nt)
#pragma unroll
    for (int i = 0; i < 8; ++i) {
      int row = qb + w * 16 + i + 8 * hh;
      int col = h * HD + nt * 16 + hl;
      O[(size_t)row * DIM + col] = oacc[nt][i] * inv[i];
    }
}

// ---------------------------------------------------------------------------
// Row-wise RMSNorm over DIM columns, in place, float4-vectorized.
// ---------------------------------------------------------------------------
__global__ __launch_bounds__(256)
void rmsnorm_kernel(float* __restrict__ X, const float* __restrict__ Wn, int N)
{
  __shared__ float red[256];
  const int row = blockIdx.x;
  float4* x4 = (float4*)(X + (size_t)row * N);
  const float4* w4 = (const float4*)Wn;
  const int n4 = N >> 2;  // 1024
  float s = 0.0f;
  for (int i = threadIdx.x; i < n4; i += 256) {
    float4 v = x4[i];
    s += v.x * v.x + v.y * v.y + v.z * v.z + v.w * v.w;
  }
  red[threadIdx.x] = s;
  __syncthreads();
  for (int o = 128; o > 0; o >>= 1) {
    if (threadIdx.x < o) red[threadIdx.x] += red[threadIdx.x + o];
    __syncthreads();
  }
  float r = rsqrtf(red[0] / (float)N + 1e-6f);
  for (int i = threadIdx.x; i < n4; i += 256) {
    float4 v = x4[i];
    float4 g = w4[i];
    v.x *= r * g.x; v.y *= r * g.y; v.z *= r * g.z; v.w *= r * g.w;
    x4[i] = v;
  }
}

// ---------------------------------------------------------------------------
extern "C" void kernel_launch(void* const* d_in, const int* in_sizes, int n_in,
                              void* d_out, int out_size, void* d_ws, size_t ws_size,
                              hipStream_t stream)
{
  (void)in_sizes; (void)n_in; (void)out_size; (void)ws_size;
  const float* video = (const float*)d_in[0];
  const float* text  = (const float*)d_in[1];
  const float* wq    = (const float*)d_in[2];
  const float* wk    = (const float*)d_in[3];
  const float* wv    = (const float*)d_in[4];
  const float* wo    = (const float*)d_in[5];
  const float* qn    = (const float*)d_in[6];
  const float* kn    = (const float*)d_in[7];
  const float* cwq   = (const float*)d_in[8];
  const float* cwk   = (const float*)d_in[9];
  const float* cwv   = (const float*)d_in[10];
  const float* cwo   = (const float*)d_in[11];
  const float* cqn   = (const float*)d_in[12];
  const float* ckn   = (const float*)d_in[13];
  const float* wffp  = (const float*)d_in[14];
  const float* wffd  = (const float*)d_in[15];
  float* out = (float*)d_out;

  float* ws = (float*)d_ws;
  float* Q  = ws;
  float* K  = Q  + (size_t)SEQ * DIM;
  float* V  = K  + (size_t)SEQ * DIM;
  float* O  = V  + (size_t)SEQ * DIM;
  float* Hb = O  + (size_t)SEQ * DIM;
  float* G  = Hb + (size_t)SEQ * DIM;   // SEQ x FFN

  dim3 blk(256);
  dim3 g_nn(DIM / 128, SEQ / 128);    // 32 x 32
  dim3 g_tn(DIM / 128, TSEQ / 128);   // 32 x 2
  dim3 g_fa(SEQ / 64, HEADS);
  dim3 blk_fa(128);

  // ---- self attention ----
  gemm_bf16<0><<<g_nn, blk, 0, stream>>>(video, wq, nullptr, Q, SEQ, DIM, DIM);
  rmsnorm_kernel<<<dim3(SEQ), blk, 0, stream>>>(Q, qn, DIM);
  gemm_bf16<0><<<g_nn, blk, 0, stream>>>(video, wk, nullptr, K, SEQ, DIM, DIM);
  rmsnorm_kernel<<<dim3(SEQ), blk, 0, stream>>>(K, kn, DIM);
  gemm_bf16<0><<<g_nn, blk, 0, stream>>>(video, wv, nullptr, V, SEQ, DIM, DIM);
  flash_attn<<<g_fa, blk_fa, 0, stream>>>(Q, K, V, O, SEQ);
  gemm_bf16<1><<<g_nn, blk, 0, stream>>>(O, wo, video, Hb, SEQ, DIM, DIM);

  // ---- cross attention (kv from text, len 256) ----
  gemm_bf16<0><<<g_nn, blk, 0, stream>>>(Hb, cwq, nullptr, Q, SEQ, DIM, DIM);
  rmsnorm_kernel<<<dim3(SEQ), blk, 0, stream>>>(Q, cqn, DIM);
  gemm_bf16<0><<<g_tn, blk, 0, stream>>>(text, cwk, nullptr, K, TSEQ, DIM, DIM);
  rmsnorm_kernel<<<dim3(TSEQ), blk, 0, stream>>>(K, ckn, DIM);
  gemm_bf16<0><<<g_tn, blk, 0, stream>>>(text, cwv, nullptr, V, TSEQ, DIM, DIM);
  flash_attn<<<g_fa, blk_fa, 0, stream>>>(Q, K, V, O, TSEQ);
  gemm_bf16<1><<<g_nn, blk, 0, stream>>>(O, cwo, Hb, Hb, SEQ, DIM, DIM);

  // ---- GEGLU FFN ----
  // proj: M on fast grid axis so weight col-blocks are shared across
  // concurrent blocks (512MB W streams once; activations L2-resident)
  gemm_geglu<<<dim3(SEQ / 128, FFN / 64), blk, 0, stream>>>(Hb, wffp, G, SEQ, DIM);
  // down: W (256MB) fits L2; stream G once with N on fast axis
  gemm_bf16<1><<<g_nn, blk, 0, stream>>>(G, wffd, Hb, out, SEQ, DIM, FFN);
}
